// GlobalAttention_14267881357662
// MI455X (gfx1250) — matrix operand order repaired and checked by reading, not compiled, not run on hardware
//
#include <hip/hip_runtime.h>
#include <hip/hip_bf16.h>

typedef _Float16 v8h  __attribute__((ext_vector_type(8)));
typedef _Float16 v16h __attribute__((ext_vector_type(16)));
typedef float    v8f  __attribute__((ext_vector_type(8)));
typedef int      v4i  __attribute__((ext_vector_type(4)));

#define Bb 4
#define Nn 2048
#define Dd 1024
#define Hh 16
#define HD 64

static __device__ __forceinline__ v16h combine8(v8h lo, v8h hi) {
    v16h r;
#pragma unroll
    for (int i = 0; i < 8; ++i) { r[i] = lo[i]; r[i + 8] = hi[i]; }
    return r;
}

static __device__ __forceinline__ v8f wmma16(v16h a, v16h b, v8f c) {
    return __builtin_amdgcn_wmma_f32_16x16x32_f16(false, a, false, b, (short)0, c, false, false);
}

// Async global->LDS 128-bit copy (gfx1250 GLOBAL_LOAD_ASYNC_TO_LDS_B128),
// with a synchronous fallback if the builtin is unavailable.
// Builtin signature (from hipcc diagnostic): (AS1 v4i* src, AS3 v4i* dst, Ii, Ii)
static __device__ __forceinline__ void async_copy_b128(const _Float16* g, _Float16* l) {
#if defined(__gfx1250__) && __has_builtin(__builtin_amdgcn_global_load_async_to_lds_b128)
    __builtin_amdgcn_global_load_async_to_lds_b128(
        (__attribute__((address_space(1))) v4i*)g,
        (__attribute__((address_space(3))) v4i*)l, 0, 0);
#else
    *(v8h*)l = *(const v8h*)g;
#endif
}

static __device__ __forceinline__ void async_wait0() {
#if defined(__gfx1250__) && __has_builtin(__builtin_amdgcn_s_wait_asynccnt)
    __builtin_amdgcn_s_wait_asynccnt(0);
#elif defined(__gfx1250__)
    asm volatile("s_wait_asynccnt 0x0" ::: "memory");
#endif
}

// ---------------------------------------------------------------- converts
__global__ void cvt_f32_f16_kernel(const float* __restrict__ src,
                                   _Float16* __restrict__ dst, int n) {
    int i = blockIdx.x * blockDim.x + threadIdx.x;
    int stride = gridDim.x * blockDim.x;
    for (; i < n; i += stride) dst[i] = (_Float16)src[i];
}

// src [rows, cols] f32 -> dst [cols, rows] f16  (weights: [K,N] -> [N,K])
__global__ void cvt_transpose_f16_kernel(const float* __restrict__ src,
                                         _Float16* __restrict__ dst,
                                         int rows, int cols) {
    int i = blockIdx.x * blockDim.x + threadIdx.x;
    int stride = gridDim.x * blockDim.x;
    int total = rows * cols;
    for (; i < total; i += stride) {
        int r = i / cols, c = i - r * cols;
        dst[(size_t)c * rows + r] = (_Float16)src[i];
    }
}

// ---------------------------------------------------------------- QKV GEMM
// A = xh [8192,1024] f16 row-major, Bt = WqkvT [3072,1024] f16 (pre-transposed).
// Output scattered to Qh[B,H,N,64], Kh[B,H,N,64], Vt[B,H,64,N] (V transposed).
__global__ void __launch_bounds__(256)
qkv_gemm_kernel(const _Float16* __restrict__ A, const _Float16* __restrict__ Bt,
                _Float16* __restrict__ Qh, _Float16* __restrict__ Kh,
                _Float16* __restrict__ Vt) {
    __shared__ __align__(16) _Float16 As[128 * 40];
    __shared__ __align__(16) _Float16 Bs[128 * 40];

    const int tileM = blockIdx.x * 128;
    const int tileN = blockIdx.y * 128;
    const int tid  = threadIdx.x;
    const int wave = tid >> 5;
    const int lane = tid & 31;
    const int g    = lane >> 4;   // half-wave group
    const int ln   = lane & 15;
    const int wm = (wave >> 2) * 64;   // 2x4 wave grid -> 64x32 per wave
    const int wn = (wave & 3) * 32;

    v8f acc[4][2];
#pragma unroll
    for (int i = 0; i < 4; ++i)
#pragma unroll
        for (int j = 0; j < 2; ++j) acc[i][j] = (v8f){};

    for (int k0 = 0; k0 < 1024; k0 += 32) {
        // async tile copies: both tiles are 128 rows x 32 halves, b128 chunks
#pragma unroll
        for (int ch = 0; ch < 2; ++ch) {
            int idx = tid + ch * 256;
            int r = idx >> 2, c8 = idx & 3;
            async_copy_b128(A  + (size_t)(tileM + r) * 1024 + k0 + c8 * 8,
                            &As[r * 40 + c8 * 8]);
            async_copy_b128(Bt + (size_t)(tileN + r) * 1024 + k0 + c8 * 8,
                            &Bs[r * 40 + c8 * 8]);
        }
        async_wait0();
        __syncthreads();

        v16h af[4], bf[2];
#pragma unroll
        for (int ms = 0; ms < 4; ++ms) {
            const _Float16* p = &As[(wm + ms * 16 + ln) * 40];
            af[ms] = combine8(*(const v8h*)(p + g * 8), *(const v8h*)(p + 16 + g * 8));
        }
#pragma unroll
        for (int ns = 0; ns < 2; ++ns) {
            const _Float16* p = &Bs[(wn + ns * 16 + ln) * 40];
            bf[ns] = combine8(*(const v8h*)(p + g * 8), *(const v8h*)(p + 16 + g * 8));
        }
#pragma unroll
        for (int ms = 0; ms < 4; ++ms)
#pragma unroll
            for (int ns = 0; ns < 2; ++ns)
                acc[ms][ns] = wmma16(af[ms], bf[ns], acc[ms][ns]);
        __syncthreads();
    }

    // epilogue: scatter into Q / K / Vt, f32->f16
#pragma unroll
    for (int ms = 0; ms < 4; ++ms)
#pragma unroll
        for (int ns = 0; ns < 2; ++ns)
#pragma unroll
            for (int e = 0; e < 8; ++e) {
                int gm = tileM + wm + ms * 16 + e + 8 * g;
                int gn = tileN + wn + ns * 16 + ln;
                _Float16 hv = (_Float16)acc[ms][ns][e];
                int b = gm >> 11, n = gm & 2047;
                if (gn < 1024) {
                    int h = gn >> 6, dd = gn & 63;
                    Qh[(((size_t)b * Hh + h) * Nn + n) * HD + dd] = hv;
                } else if (gn < 2048) {
                    int j = gn - 1024, h = j >> 6, dd = j & 63;
                    Kh[(((size_t)b * Hh + h) * Nn + n) * HD + dd] = hv;
                } else {
                    int j = gn - 2048, h = j >> 6, dd = j & 63;
                    Vt[(((size_t)b * Hh + h) * HD + dd) * Nn + n] = hv;
                }
            }
}

// ---------------------------------------------------------------- attention
// One workgroup (8 waves) per 16-query stripe of one (b,h).
// LDS: S f32[16][2048] (128KB) + P f16[16][2048] (64KB); S region reused for
// the 8x16x64 f32 partial-output reduction.
__global__ void __launch_bounds__(256)
attn_kernel(const _Float16* __restrict__ Qh, const _Float16* __restrict__ Kh,
            const _Float16* __restrict__ Vt, _Float16* __restrict__ ATT) {
    extern __shared__ char smem[];
    float*    S = (float*)smem;                        // [16][2048]
    _Float16* P = (_Float16*)(smem + 16 * 2048 * 4);   // [16][2048]
    float* Opart = (float*)smem;                       // [8][16][64] (reuse)

    const int id = blockIdx.x;
    const int qt = id & 127;            // N/16 = 128 stripes
    const int bh = id >> 7;             // b*16 + h
    const int qbase = qt * 16;
    const int tid  = threadIdx.x;
    const int wave = tid >> 5;
    const int lane = tid & 31;
    const int g    = lane >> 4;
    const int ln   = lane & 15;
    const float scale = 0.125f;         // 1/sqrt(64)

    // Q fragments (A-operand layout), loaded once per wave
    const _Float16* qptr = Qh + ((size_t)bh * Nn + qbase + ln) * HD;
    v16h a0 = combine8(*(const v8h*)(qptr + g * 8),      *(const v8h*)(qptr + 16 + g * 8));
    v16h a1 = combine8(*(const v8h*)(qptr + 32 + g * 8), *(const v8h*)(qptr + 48 + g * 8));

    // ---- phase 1: S = Q K^T (16 x 2048), waves strided over key tiles
    for (int kt = wave; kt < 128; kt += 8) {
        const int kbase = kt * 16;
        const _Float16* kptr = Kh + ((size_t)bh * Nn + kbase + ln) * HD;
        v16h b0 = combine8(*(const v8h*)(kptr + g * 8),      *(const v8h*)(kptr + 16 + g * 8));
        v16h b1 = combine8(*(const v8h*)(kptr + 32 + g * 8), *(const v8h*)(kptr + 48 + g * 8));
        v8f c = (v8f){};
        c = wmma16(a0, b0, c);
        c = wmma16(a1, b1, c);
#pragma unroll
        for (int e = 0; e < 8; ++e)
            S[(e + 8 * g) * 2048 + kbase + ln] = c[e];
    }
    __syncthreads();

    // ---- phase 2: exact softmax per row (16 threads per row, same half-wave)
    {
        const int row = tid >> 4, li = tid & 15;
        float* srow = S + row * 2048;
        float mx = -1e30f;
        for (int i = li; i < 2048; i += 16) mx = fmaxf(mx, srow[i] * scale);
#pragma unroll
        for (int m = 1; m < 16; m <<= 1) mx = fmaxf(mx, __shfl_xor(mx, m, 32));
        float sum = 0.0f;
        for (int i = li; i < 2048; i += 16) {
            float p = __expf(srow[i] * scale - mx);
            srow[i] = p;
            sum += p;
        }
#pragma unroll
        for (int m = 1; m < 16; m <<= 1) sum += __shfl_xor(sum, m, 32);
        float inv = 1.0f / sum;
        _Float16* prow = P + row * 2048;
        for (int i = li; i < 2048; i += 16) prow[i] = (_Float16)(srow[i] * inv);
    }
    __syncthreads();

    // ---- phase 3: O = P @ V, contraction split 256 keys per wave
    v8f o[4];
#pragma unroll
    for (int nt = 0; nt < 4; ++nt) o[nt] = (v8f){};
    for (int it = 0; it < 8; ++it) {
        const int kb = wave * 256 + it * 32;
        const _Float16* pb = P + ln * 2048 + kb;
        v16h pa = combine8(*(const v8h*)(pb + g * 8), *(const v8h*)(pb + 16 + g * 8));
#pragma unroll
        for (int nt = 0; nt < 4; ++nt) {
            const _Float16* vptr =
                Vt + ((size_t)bh * HD + nt * 16 + ln) * Nn + kb + g * 8;
            v16h vb = combine8(*(const v8h*)vptr, *(const v8h*)(vptr + 16));
            o[nt] = wmma16(pa, vb, o[nt]);
        }
    }
    // stash partials into reused S region (safe: S last read before barrier)
#pragma unroll
    for (int nt = 0; nt < 4; ++nt)
#pragma unroll
        for (int e = 0; e < 8; ++e)
            Opart[(wave * 16 + e + 8 * g) * 64 + nt * 16 + ln] = o[nt][e];
    __syncthreads();

    // ---- phase 4: reduce 8 partials, emit context into ATT [8192,1024] f16
    const int b = bh >> 4, h = bh & 15;
    for (int i = tid; i < 1024; i += 256) {
        int r = i >> 6, d = i & 63;
        float accv = 0.0f;
#pragma unroll
        for (int w = 0; w < 8; ++w) accv += Opart[(w * 16 + r) * 64 + d];
        ATT[((size_t)b * Nn + qbase + r) * Dd + h * HD + d] = (_Float16)accv;
    }
}

// ---------------------------------------------------------------- out GEMM
// out[8192,1024] = ATT[8192,1024] @ Wout, Bt = WoutT [1024,1024] pre-transposed.
// Masked f32 epilogue.
__global__ void __launch_bounds__(256)
out_gemm_kernel(const _Float16* __restrict__ A, const _Float16* __restrict__ Bt,
                const unsigned char* __restrict__ mask, float* __restrict__ out) {
    __shared__ __align__(16) _Float16 As[128 * 40];
    __shared__ __align__(16) _Float16 Bs[128 * 40];

    const int tileM = blockIdx.x * 128;
    const int tileN = blockIdx.y * 128;
    const int tid  = threadIdx.x;
    const int wave = tid >> 5;
    const int lane = tid & 31;
    const int g    = lane >> 4;
    const int ln   = lane & 15;
    const int wm = (wave >> 2) * 64;
    const int wn = (wave & 3) * 32;

    v8f acc[4][2];
#pragma unroll
    for (int i = 0; i < 4; ++i)
#pragma unroll
        for (int j = 0; j < 2; ++j) acc[i][j] = (v8f){};

    for (int k0 = 0; k0 < 1024; k0 += 32) {
#pragma unroll
        for (int ch = 0; ch < 2; ++ch) {
            int idx = tid + ch * 256;
            int r = idx >> 2, c8 = idx & 3;
            async_copy_b128(A  + (size_t)(tileM + r) * 1024 + k0 + c8 * 8,
                            &As[r * 40 + c8 * 8]);
            async_copy_b128(Bt + (size_t)(tileN + r) * 1024 + k0 + c8 * 8,
                            &Bs[r * 40 + c8 * 8]);
        }
        async_wait0();
        __syncthreads();

        v16h af[4], bf[2];
#pragma unroll
        for (int ms = 0; ms < 4; ++ms) {
            const _Float16* p = &As[(wm + ms * 16 + ln) * 40];
            af[ms] = combine8(*(const v8h*)(p + g * 8), *(const v8h*)(p + 16 + g * 8));
        }
#pragma unroll
        for (int ns = 0; ns < 2; ++ns) {
            const _Float16* p = &Bs[(wn + ns * 16 + ln) * 40];
            bf[ns] = combine8(*(const v8h*)(p + g * 8), *(const v8h*)(p + 16 + g * 8));
        }
#pragma unroll
        for (int ms = 0; ms < 4; ++ms)
#pragma unroll
            for (int ns = 0; ns < 2; ++ns)
                acc[ms][ns] = wmma16(af[ms], bf[ns], acc[ms][ns]);
        __syncthreads();
    }

#pragma unroll
    for (int ms = 0; ms < 4; ++ms)
#pragma unroll
        for (int ns = 0; ns < 2; ++ns)
#pragma unroll
            for (int e = 0; e < 8; ++e) {
                int gm = tileM + wm + ms * 16 + e + 8 * g;
                int gn = tileN + wn + ns * 16 + ln;
                int b = gm >> 11, n = gm & 2047;
                float v = mask[b * Nn + n] ? acc[ms][ns][e] : 0.0f;
                out[(size_t)gm * Dd + gn] = v;
            }
}

// ---------------------------------------------------------------- launch
extern "C" void kernel_launch(void* const* d_in, const int* in_sizes, int n_in,
                              void* d_out, int out_size, void* d_ws, size_t ws_size,
                              hipStream_t stream) {
    (void)in_sizes; (void)n_in; (void)out_size; (void)ws_size;
    const float* x            = (const float*)d_in[0];
    const unsigned char* mask = (const unsigned char*)d_in[1];
    const float* Wqkv         = (const float*)d_in[2];
    const float* Wout         = (const float*)d_in[3];
    float* out = (float*)d_out;

    char* ws = (char*)d_ws;
    _Float16* xh     = (_Float16*)(ws);                          // 16 MiB
    _Float16* wqkvT  = (_Float16*)(ws + (size_t)16 * 1048576);   //  6 MiB [3072,1024]
    _Float16* woutT  = (_Float16*)(ws + (size_t)22 * 1048576);   //  2 MiB [1024,1024]
    _Float16* Qh     = (_Float16*)(ws + (size_t)24 * 1048576);   // 16 MiB
    _Float16* Kh     = (_Float16*)(ws + (size_t)40 * 1048576);   // 16 MiB
    _Float16* Vt     = (_Float16*)(ws + (size_t)56 * 1048576);   // 16 MiB
    _Float16* ATT    = (_Float16*)(ws + (size_t)72 * 1048576);   // 16 MiB

    cvt_f32_f16_kernel<<<4096, 256, 0, stream>>>(x, xh, Bb * Nn * Dd);
    cvt_transpose_f16_kernel<<<2048, 256, 0, stream>>>(Wqkv, wqkvT, Dd, 3 * Dd);
    cvt_transpose_f16_kernel<<<1024, 256, 0, stream>>>(Wout, woutT, Dd, Dd);

    qkv_gemm_kernel<<<dim3(64, 24), 256, 0, stream>>>(xh, wqkvT, Qh, Kh, Vt);

    (void)hipFuncSetAttribute(reinterpret_cast<const void*>(attn_kernel),
                              hipFuncAttributeMaxDynamicSharedMemorySize, 196608);
    attn_kernel<<<Bb * Hh * (Nn / 16), 256, 196608, stream>>>(Qh, Kh, Vt, ATT);

    out_gemm_kernel<<<dim3(64, 8), 256, 0, stream>>>(ATT, woutT, mask, out);
}
